// PeriodicSetTransformer_47467978556071
// MI455X (gfx1250) — matrix operand order repaired and checked by reading, not compiled, not run on hardware
//
#include <hip/hip_runtime.h>
#include <hip/hip_bf16.h>
#include <math.h>

// ---------------------------------------------------------------------------
// PeriodicSetTransformer forward, MI455X (gfx1250), wave32 + WMMA bf16,
// LDS double-buffered tiles with async global->LDS DMA (ASYNCcnt) when the
// toolchain exposes the gfx1250 builtins.
// B=32 N=512 Kin=101 E=128 H=8 L=3 EH=1024
// ---------------------------------------------------------------------------

#define B_     32
#define N_     512
#define K_IN   101
#define E_     128
#define H_     8
#define L_     3
#define EH_    1024
#define A_DIM_ 200
#define BN_    (B_ * N_)   // 16384

typedef __attribute__((ext_vector_type(16))) __bf16         v16bf;
typedef __attribute__((ext_vector_type(8)))  float          v8f;
typedef __attribute__((ext_vector_type(8)))  unsigned short v8u;
typedef __attribute__((ext_vector_type(16))) unsigned short v16u;
typedef __attribute__((ext_vector_type(4)))  int            v4i_;

typedef __attribute__((address_space(1))) v4i_* gv4i_p;   // global 16B vector ptr
typedef __attribute__((address_space(3))) v4i_* lv4i_p;   // LDS    16B vector ptr

#if defined(__has_builtin)
#if __has_builtin(__builtin_amdgcn_global_load_async_to_lds_b128)
#define HAVE_ASYNC_LDS 1
#else
#define HAVE_ASYNC_LDS 0
#endif
#else
#define HAVE_ASYNC_LDS 0
#endif

__device__ __forceinline__ unsigned short f2bf(float f) {
  unsigned int u = __float_as_uint(f);
  u += 0x7FFFu + ((u >> 16) & 1u);          // round-to-nearest-even
  return (unsigned short)(u >> 16);
}

__device__ __forceinline__ float mishf(float v) {
  float sp = (v > 20.f) ? v : log1pf(__expf(v));
  return v * tanhf(sp);
}

__device__ __forceinline__ v16bf make_frag(v8u lo, v8u hi) {
  v16u t = __builtin_shufflevector(lo, hi, 0,1,2,3,4,5,6,7,8,9,10,11,12,13,14,15);
  return __builtin_bit_cast(v16bf, t);
}

// 16-byte global -> LDS copy; async DMA on gfx1250 toolchains that expose it.
__device__ __forceinline__ void g2l16(unsigned short* l, const unsigned short* g) {
#if HAVE_ASYNC_LDS
  __builtin_amdgcn_global_load_async_to_lds_b128((gv4i_p)g, (lv4i_p)l, 0, 0);
#else
  *(v8u*)l = *(const v8u*)g;
#endif
}

__device__ __forceinline__ void async_wait() {
#if HAVE_ASYNC_LDS
#if __has_builtin(__builtin_amdgcn_s_wait_asynccnt)
  __builtin_amdgcn_s_wait_asynccnt(0);
#else
  asm volatile("s_wait_asynccnt 0x0" ::: "memory");
#endif
#endif
}

// A fragment (16x32 bf16): elems 0..7 = K[half*8..], elems 8..15 = +16.
__device__ __forceinline__ v16bf fragA_lds(const unsigned short* base) {
  v8u lo = *(const v8u*)(base);
  v8u hi = *(const v8u*)(base + 16);
  return make_frag(lo, hi);
}
// B fragment (32x16 bf16) from K-major row: elems 0..7 = K[half*8..], elems 8..15 = +16.
__device__ __forceinline__ v16bf fragB_lds(const unsigned short* base) {
  v8u lo = *(const v8u*)(base);
  v8u hi = *(const v8u*)(base + 16);
  return make_frag(lo, hi);
}
// Direct-from-global variant (used for per-wave Q fragments).
__device__ __forceinline__ v16bf load_fragA(const unsigned short* A, size_t lda,
                                            int m0, int k0, int lane) {
  int row  = m0 + (lane & 15);
  int half = lane >> 4;
  return fragA_lds(A + (size_t)row * lda + (size_t)(k0 + half * 8));
}

__device__ __forceinline__ v8f wmma_bf16(v16bf a, v16bf b, v8f c) {
  return __builtin_amdgcn_wmma_f32_16x16x32_bf16(false, a, false, b,
                                                 (short)0, c, false, false);
}

// ---------------------------------------------------------------------------
// Embedding: x = atom_table[comp_fea] @ comp_W + str_fea[:,:,1:] @ pdd_W + b
// ---------------------------------------------------------------------------
__global__ void embed_kernel(const float* __restrict__ str_fea,
                             const int*   __restrict__ comp_fea,
                             const float* __restrict__ atom_table,
                             const float* __restrict__ comp_W,
                             const float* __restrict__ comp_b,
                             const float* __restrict__ pdd_W,
                             const float* __restrict__ pdd_b,
                             float* __restrict__ x,
                             float* __restrict__ xi,
                             float* __restrict__ wts) {
  const int bn = blockIdx.x;           // 0..BN_-1
  const int t  = threadIdx.x;          // 0..127
  __shared__ float arow[A_DIM_];
  __shared__ float srow[K_IN - 1];
  const float* sf = str_fea + (size_t)bn * K_IN;
  const int aidx  = comp_fea[bn];
  for (int i = t; i < A_DIM_;   i += 128) arow[i] = atom_table[(size_t)aidx * A_DIM_ + i];
  for (int i = t; i < K_IN - 1; i += 128) srow[i] = sf[1 + i];
  if (t == 0) wts[bn] = sf[0];
  __syncthreads();
  float acc = comp_b[t] + pdd_b[t];
  for (int a = 0; a < A_DIM_;   ++a) acc += arow[a] * comp_W[(size_t)a * E_ + t];
  for (int j = 0; j < K_IN - 1; ++j) acc += srow[j] * pdd_W[(size_t)j * E_ + t];
  x [(size_t)bn * E_ + t] = acc;
  xi[(size_t)bn * E_ + t] = acc;
}

// ---------------------------------------------------------------------------
// fp32 (Kd x Nd) weight -> bf16 transposed (Nd x Kd)
// ---------------------------------------------------------------------------
__global__ void convT_kernel(const float* __restrict__ src,
                             unsigned short* __restrict__ dst, int Kd, int Nd) {
  int i = blockIdx.x * 256 + threadIdx.x;
  if (i < Kd * Nd) {
    int k = i / Nd, n = i - k * Nd;
    dst[(size_t)n * Kd + k] = f2bf(src[i]);
  }
}

// ---------------------------------------------------------------------------
// LayerNorm over E=128; one wave per row; optional fp32 and/or bf16 output
// ---------------------------------------------------------------------------
__global__ void ln_kernel(const float* __restrict__ in,
                          const float* __restrict__ g,
                          const float* __restrict__ bb,
                          float* __restrict__ outF,
                          unsigned short* __restrict__ outU) {
  const int lane = threadIdx.x & 31;
  const int wave = threadIdx.x >> 5;
  const size_t row = (size_t)blockIdx.x * 8 + wave;
  const float4 xv = *((const float4*)(in + row * E_) + lane);
  float s = xv.x + xv.y + xv.z + xv.w;
#pragma unroll
  for (int i = 1; i < 32; i <<= 1) s += __shfl_xor(s, i, 32);
  float mean = s * (1.f / 128.f);
  float d0 = xv.x - mean, d1 = xv.y - mean, d2 = xv.z - mean, d3 = xv.w - mean;
  float vs = d0 * d0 + d1 * d1 + d2 * d2 + d3 * d3;
#pragma unroll
  for (int i = 1; i < 32; i <<= 1) vs += __shfl_xor(vs, i, 32);
  float rstd = rsqrtf(vs * (1.f / 128.f) + 1e-5f);
  float dv[4] = {d0, d1, d2, d3};
#pragma unroll
  for (int j = 0; j < 4; ++j) {
    int idx = lane * 4 + j;
    float val = dv[j] * rstd * g[idx] + bb[idx];
    if (outF) outF[row * E_ + idx] = val;
    if (outU) outU[row * E_ + idx] = f2bf(val);
  }
}

// ---------------------------------------------------------------------------
// WMMA GEMM, LDS double-buffered:  C[M,N] = A[M,K](bf16) * BT[N,K]^T + bias
// block = 256 threads (8 waves); block tile 128(M) x 64(N), K-step 32.
// mode 0: scatter to Q / K / V^T   mode 1: store bf16
// mode 2: outF = addSrc + val      mode 3: outF = addSrc + mish(val)
// ---------------------------------------------------------------------------
__global__ void gemm_bf16_wmma(const unsigned short* __restrict__ A,
                               const unsigned short* __restrict__ BT,
                               const float* __restrict__ bias,
                               int N, int K, int mode,
                               float* __restrict__ outF,
                               unsigned short* __restrict__ outU,
                               const float* __restrict__ addSrc,
                               unsigned short* __restrict__ qO,
                               unsigned short* __restrict__ kO,
                               unsigned short* __restrict__ vO) {
  __shared__ __attribute__((aligned(32))) unsigned short sA[2][128 * 32]; // 2x8KB
  __shared__ __attribute__((aligned(32))) unsigned short sB[2][64 * 32];  // 2x4KB
  const int tid  = threadIdx.x;
  const int lane = tid & 31;
  const int wave = tid >> 5;
  const int m0 = blockIdx.x * 128;
  const int n0 = blockIdx.y * 64;
  const int half = lane >> 4;
  const int colw = lane & 15;

  auto stage = [&](int k0, int buf) {
    // A tile: 128 rows x 32 cols = 512 x 16B chunks, 2 per thread
#pragma unroll
    for (int c = tid; c < 512; c += 256) {
      int row = c >> 2, part = c & 3;
      g2l16(&sA[buf][row * 32 + part * 8],
            A + (size_t)(m0 + row) * K + (size_t)(k0 + part * 8));
    }
    // B tile: 64 rows x 32 cols = 256 x 16B chunks, 1 per thread
    {
      int row = tid >> 2, part = tid & 3;
      g2l16(&sB[buf][row * 32 + part * 8],
            BT + (size_t)(n0 + row) * K + (size_t)(k0 + part * 8));
    }
  };

  v8f acc[4] = {};
  stage(0, 0);
  async_wait();
  __syncthreads();

  const int arow = wave * 16 + colw;           // A-frag row within block tile
  for (int k0 = 0; k0 < K; k0 += 32) {
    int cur = (k0 >> 5) & 1;
    if (k0 + 32 < K) stage(k0 + 32, cur ^ 1);  // overlap DMA with compute
    v16bf a = fragA_lds(&sA[cur][arow * 32 + half * 8]);
#pragma unroll
    for (int t = 0; t < 4; ++t) {
      v16bf b = fragB_lds(&sB[cur][(t * 16 + colw) * 32 + 8 * half]);
      acc[t] = wmma_bf16(a, b, acc[t]);
    }
    if (k0 + 32 < K) { async_wait(); __syncthreads(); }
  }

#pragma unroll
  for (int t = 0; t < 4; ++t) {
    int col = n0 + t * 16 + colw;
    float bv = bias[col];
#pragma unroll
    for (int v = 0; v < 8; ++v) {
      int row = m0 + wave * 16 + v + 8 * half;   // C layout: row = v + 8*half
      float val = acc[t][v] + bv;
      if (mode == 0) {
        int h  = col / (3 * E_);
        int rr = col - h * (3 * E_);
        int bb = row >> 9;                // / N_
        int n  = row & (N_ - 1);
        int bh = bb * H_ + h;
        if (rr < E_)
          qO[((size_t)(bh * N_ + n)) * E_ + rr] = f2bf(val);
        else if (rr < 2 * E_)
          kO[((size_t)(bh * N_ + n)) * E_ + (rr - E_)] = f2bf(val);
        else
          vO[((size_t)(bh * E_ + (rr - 2 * E_))) * N_ + n] = f2bf(val);
      } else if (mode == 1) {
        outU[(size_t)row * N + col] = f2bf(val);
      } else if (mode == 2) {
        outF[(size_t)row * N + col] = addSrc[(size_t)row * N + col] + val;
      } else {
        outF[(size_t)row * N + col] = addSrc[(size_t)row * N + col] + mishf(val);
      }
    }
  }
}

// ---------------------------------------------------------------------------
// Fused flash attention (masked, weights-multiplied softmax), WMMA bf16.
// 4 waves/block share LDS-staged, double-buffered K/V tiles; one wave owns
// 16 query rows of one (b,h).
// ---------------------------------------------------------------------------
__global__ void attn_wmma(const unsigned short* __restrict__ Qb,
                          const unsigned short* __restrict__ Kb,
                          const unsigned short* __restrict__ Vt,
                          const float* __restrict__ wts,
                          unsigned short* __restrict__ vals) {
  __shared__ __attribute__((aligned(32))) unsigned short sK[2][32 * 128];  // 2x8KB
  __shared__ __attribute__((aligned(32))) unsigned short sV[2][128 * 32];  // 2x8KB
  __shared__ __attribute__((aligned(32))) unsigned short pbuf[4][16 * 32]; // 4KB
  const int tid  = threadIdx.x;
  const int lane = tid & 31;
  const int wave = tid >> 5;
  const int bh = blockIdx.y;
  const int b  = bh >> 3;          // H_ = 8
  const int h  = bh & 7;
  const int q0 = (blockIdx.x * 4 + wave) * 16;

  const unsigned short* Q  = Qb + (size_t)bh * N_ * E_;
  const unsigned short* Kp = Kb + (size_t)bh * N_ * E_;
  const unsigned short* Vp = Vt + (size_t)bh * E_ * N_;   // (E x N) = V^T
  const float* wrow = wts + (size_t)b * N_;

  const int half = lane >> 4;
  const int colw = lane & 15;

  auto stage_kv = [&](int kb, int buf) {
    // K tile: 32 key rows x 128 elems = 512 x 16B chunks, 4 per thread
#pragma unroll
    for (int c = tid; c < 512; c += 128) {
      int row = c >> 4, part = c & 15;
      g2l16(&sK[buf][row * 128 + part * 8],
            Kp + (size_t)(kb + row) * E_ + (size_t)(part * 8));
    }
    // V^T tile: 128 e-rows x 32 keys = 512 x 16B chunks, 4 per thread
#pragma unroll
    for (int c = tid; c < 512; c += 128) {
      int row = c >> 2, part = c & 3;
      g2l16(&sV[buf][row * 32 + part * 8],
            Vp + (size_t)row * N_ + (size_t)(kb + part * 8));
    }
  };

  v16bf qa[4];
#pragma unroll
  for (int e = 0; e < 4; ++e) qa[e] = load_fragA(Q, E_, q0, e * 32, lane);

  float mrow[8], srow[8];
  bool  keepq[8];
  v8f   o[8] = {};
#pragma unroll
  for (int v = 0; v < 8; ++v) {
    mrow[v]  = -3.0e38f;
    srow[v]  = 0.f;
    keepq[v] = wrow[q0 + v + 8 * half] > 0.f;
  }
  const float scale = 0.08838834764831845f;   // 1/sqrt(128)

  stage_kv(0, 0);
  async_wait();
  __syncthreads();

  for (int kb = 0; kb < N_; kb += 32) {
    int cur = (kb >> 5) & 1;
    if (kb + 32 < N_) stage_kv(kb + 32, cur ^ 1);

    v8f l0f = {}, l1f = {};
#pragma unroll
    for (int e = 0; e < 4; ++e) {
      v16bf b0 = fragB_lds(&sK[cur][(colw)      * 128 + e * 32 + 8 * half]);
      v16bf b1 = fragB_lds(&sK[cur][(16 + colw) * 128 + e * 32 + 8 * half]);
      l0f = wmma_bf16(qa[e], b0, l0f);
      l1f = wmma_bf16(qa[e], b1, l1f);
    }
    float w0 = wrow[kb + colw];
    float w1 = wrow[kb + 16 + colw];
    bool k0ok = w0 > 0.f, k1ok = w1 > 0.f;
    unsigned short* pb = pbuf[wave];
#pragma unroll
    for (int v = 0; v < 8; ++v) {
      float l0 = (keepq[v] && k0ok) ? l0f[v] * scale : -9.0e15f;
      float l1 = (keepq[v] && k1ok) ? l1f[v] * scale : -9.0e15f;
      float mx = fmaxf(l0, l1);
#pragma unroll
      for (int i = 1; i < 16; i <<= 1) mx = fmaxf(mx, __shfl_xor(mx, i, 32));
      float mnew  = fmaxf(mrow[v], mx);
      float alpha = __expf(mrow[v] - mnew);
      float p0 = w0 * __expf(l0 - mnew);
      float p1 = w1 * __expf(l1 - mnew);
      float rs = p0 + p1;
#pragma unroll
      for (int i = 1; i < 16; i <<= 1) rs += __shfl_xor(rs, i, 32);
      srow[v] = srow[v] * alpha + rs;
      mrow[v] = mnew;
#pragma unroll
      for (int e = 0; e < 8; ++e) o[e][v] *= alpha;
      int row = v + 8 * half;
      pb[row * 32 + colw]      = f2bf(p0);
      pb[row * 32 + 16 + colw] = f2bf(p1);
    }
    // Re-read P tile from LDS in A-operand layout (16x32, K = key-in-block).
    v16bf pa = fragA_lds(pb + (size_t)colw * 32 + half * 8);
#pragma unroll
    for (int e = 0; e < 8; ++e) {
      v16bf vb = fragB_lds(&sV[cur][(e * 16 + colw) * 32 + 8 * half]);
      o[e] = wmma_bf16(pa, vb, o[e]);
    }
    if (kb + 32 < N_) { async_wait(); __syncthreads(); }
  }

#pragma unroll
  for (int v = 0; v < 8; ++v) {
    float inv = 1.f / srow[v];
    int n = q0 + v + 8 * half;
#pragma unroll
    for (int e = 0; e < 8; ++e) {
      float val = o[e][v] * inv;
      vals[((size_t)(b * N_ + n)) * EH_ + h * E_ + e * 16 + colw] = f2bf(val);
    }
  }
}

// ---------------------------------------------------------------------------
// pooled = LN( sum_n w * (x + x_init) );  out = pooled @ head_W + head_b
// ---------------------------------------------------------------------------
__global__ void pool_head_kernel(const float* __restrict__ x,
                                 const float* __restrict__ xi,
                                 const float* __restrict__ wts,
                                 const float* __restrict__ g,
                                 const float* __restrict__ bb,
                                 const float* __restrict__ hW,
                                 const float* __restrict__ hB,
                                 float* __restrict__ out) {
  const int b = blockIdx.x;
  const int t = threadIdx.x;       // 0..127
  float acc = 0.f;
  for (int n = 0; n < N_; ++n) {
    size_t base = (size_t)b * N_ + n;
    float w = wts[base];
    size_t i = base * E_ + t;
    acc += w * (x[i] + xi[i]);
  }
  __shared__ float sh[128];
  sh[t] = acc;
  __syncthreads();
  float s = 0.f;
  for (int i = 0; i < 128; ++i) s += sh[i];
  float mean = s * (1.f / 128.f);
  __syncthreads();
  float d = acc - mean;
  sh[t] = d * d;
  __syncthreads();
  float vs = 0.f;
  for (int i = 0; i < 128; ++i) vs += sh[i];
  float rstd = rsqrtf(vs * (1.f / 128.f) + 1e-5f);
  float nv = d * rstd * g[t] + bb[t];
  __syncthreads();
  sh[t] = nv * hW[t];
  __syncthreads();
  if (t == 0) {
    float r = 0.f;
    for (int i = 0; i < 128; ++i) r += sh[i];
    out[b] = r + hB[0];
  }
}

// ---------------------------------------------------------------------------
extern "C" void kernel_launch(void* const* d_in, const int* in_sizes, int n_in,
                              void* d_out, int out_size, void* d_ws, size_t ws_size,
                              hipStream_t stream) {
  (void)in_sizes; (void)n_in; (void)out_size; (void)ws_size;
  const float* str_fea    = (const float*)d_in[0];
  const int*   comp_fea   = (const int*)d_in[1];
  /* d_in[2] = cell_fea: unused by the reference */
  const float* atom_table = (const float*)d_in[3];
  const float* comp_W     = (const float*)d_in[4];
  const float* comp_b     = (const float*)d_in[5];
  const float* pdd_W      = (const float*)d_in[6];
  const float* pdd_b      = (const float*)d_in[7];
  const float* enc_ln_g   = (const float*)d_in[8];
  const float* enc_ln_b   = (const float*)d_in[9];
  const float* qkv_W      = (const float*)d_in[10];
  const float* qkv_b      = (const float*)d_in[11];
  const float* o_W        = (const float*)d_in[12];
  const float* o_b        = (const float*)d_in[13];
  const float* out_W      = (const float*)d_in[14];
  const float* out_b      = (const float*)d_in[15];
  const float* ffn_W      = (const float*)d_in[16];
  const float* ffn_b      = (const float*)d_in[17];
  const float* ln2_g      = (const float*)d_in[18];
  const float* ln2_b      = (const float*)d_in[19];
  const float* head_W     = (const float*)d_in[20];
  const float* head_b     = (const float*)d_in[21];

  char* ws = (char*)d_ws;
  size_t off = 0;
  auto carve = [&](size_t bytes) -> void* {
    void* p = ws + off;
    off = (off + bytes + 255) & ~(size_t)255;
    return p;
  };
  float* x      = (float*)carve((size_t)BN_ * E_ * 4);
  float* x_init = (float*)carve((size_t)BN_ * E_ * 4);
  float* out1   = (float*)carve((size_t)BN_ * E_ * 4);
  float* tmp    = (float*)carve((size_t)BN_ * E_ * 4);
  float* wts    = (float*)carve((size_t)BN_ * 4);
  unsigned short* xn_bf = (unsigned short*)carve((size_t)BN_ * E_ * 2);
  unsigned short* Qb    = (unsigned short*)carve((size_t)BN_ * E_ * 2);
  unsigned short* Kb    = (unsigned short*)carve((size_t)BN_ * E_ * 2);
  unsigned short* Vt    = (unsigned short*)carve((size_t)BN_ * E_ * 2);
  unsigned short* vals  = (unsigned short*)carve((size_t)BN_ * EH_ * 2);
  unsigned short* obf   = (unsigned short*)carve((size_t)BN_ * EH_ * 2);
  unsigned short* wqkvT = (unsigned short*)carve((size_t)L_ * 3 * EH_ * E_ * 2);
  unsigned short* woT   = (unsigned short*)carve((size_t)L_ * EH_ * EH_ * 2);
  unsigned short* woutT = (unsigned short*)carve((size_t)L_ * E_ * EH_ * 2);
  unsigned short* wffnT = (unsigned short*)carve((size_t)L_ * E_ * E_ * 2);

  embed_kernel<<<BN_, 128, 0, stream>>>(str_fea, comp_fea, atom_table,
                                        comp_W, comp_b, pdd_W, pdd_b,
                                        x, x_init, wts);

  for (int i = 0; i < L_; ++i) {
    convT_kernel<<<(E_ * 3 * EH_ + 255) / 256, 256, 0, stream>>>(
        qkv_W + (size_t)i * E_ * 3 * EH_, wqkvT + (size_t)i * 3 * EH_ * E_, E_, 3 * EH_);
    convT_kernel<<<(EH_ * EH_ + 255) / 256, 256, 0, stream>>>(
        o_W + (size_t)i * EH_ * EH_, woT + (size_t)i * EH_ * EH_, EH_, EH_);
    convT_kernel<<<(EH_ * E_ + 255) / 256, 256, 0, stream>>>(
        out_W + (size_t)i * EH_ * E_, woutT + (size_t)i * E_ * EH_, EH_, E_);
    convT_kernel<<<(E_ * E_ + 255) / 256, 256, 0, stream>>>(
        ffn_W + (size_t)i * E_ * E_, wffnT + (size_t)i * E_ * E_, E_, E_);
  }

  for (int i = 0; i < L_; ++i) {
    const float* gi = enc_ln_g + i * E_;
    const float* bi = enc_ln_b + i * E_;

    // xn = LN(x) -> bf16
    ln_kernel<<<BN_ / 8, 256, 0, stream>>>(x, gi, bi, nullptr, xn_bf);

    // QKV projection, scatter to Q, K, V^T
    gemm_bf16_wmma<<<dim3(BN_ / 128, 3 * EH_ / 64), 256, 0, stream>>>(
        xn_bf, wqkvT + (size_t)i * 3 * EH_ * E_, qkv_b + (size_t)i * 3 * EH_,
        3 * EH_, E_, 0, nullptr, nullptr, nullptr, Qb, Kb, Vt);

    // fused masked/weighted flash attention -> vals (bf16, B x N x EH)
    attn_wmma<<<dim3(N_ / 64, B_ * H_), 128, 0, stream>>>(Qb, Kb, Vt, wts, vals);

    // o = vals @ o_W + o_b   (bf16 out)
    gemm_bf16_wmma<<<dim3(BN_ / 128, EH_ / 64), 256, 0, stream>>>(
        vals, woT + (size_t)i * EH_ * EH_, o_b + (size_t)i * EH_,
        EH_, EH_, 1, nullptr, obf, nullptr, nullptr, nullptr, nullptr);

    // out1 = x + o @ out_W + out_b   (fp32)
    gemm_bf16_wmma<<<dim3(BN_ / 128, E_ / 64), 256, 0, stream>>>(
        obf, woutT + (size_t)i * E_ * EH_, out_b + (size_t)i * E_,
        E_, EH_, 2, out1, nullptr, x, nullptr, nullptr, nullptr);

    // LN(out1) -> bf16 (reuse xn_bf)
    ln_kernel<<<BN_ / 8, 256, 0, stream>>>(out1, gi, bi, nullptr, xn_bf);

    // tmp = out1 + mish(LN(out1) @ ffn_W + ffn_b)
    gemm_bf16_wmma<<<dim3(BN_ / 128, E_ / 64), 256, 0, stream>>>(
        xn_bf, wffnT + (size_t)i * E_ * E_, ffn_b + (size_t)i * E_,
        E_, E_, 3, tmp, nullptr, out1, nullptr, nullptr, nullptr);

    // x = LN(tmp)  (fp32)
    ln_kernel<<<BN_ / 8, 256, 0, stream>>>(tmp, gi, bi, x, nullptr);
  }

  pool_head_kernel<<<B_, 128, 0, stream>>>(x, x_init, wts, ln2_g, ln2_b,
                                           head_W, head_b, (float*)d_out);
}
